// EdgeAwareAttention_69887707840996
// MI455X (gfx1250) — compile-verified, matrix operand order checked by reading, not run on hardware
//
#include <hip/hip_runtime.h>
#include <hip/hip_bf16.h>

#define NHEADS 12
#define HD     64
#define HDIM   768
#define SEQ    1024
#define BATCH  8
#define ATT_SCALE 0.125f  // 1/sqrt(64)

typedef __attribute__((ext_vector_type(16))) __bf16 v16bf;
typedef __attribute__((ext_vector_type(8)))  __bf16 v8bf;
typedef __attribute__((ext_vector_type(8)))  float  v8f;

// A/B fragment for v_wmma_f32_16x16x32_bf16, operand rows contiguous in memory.
// Element v holds k = (v/8)*16 + hi*8 + (v%8)  (hi = lane>=16), i.e. two
// contiguous 8xbf16 (16-byte) chunks at +hi*8 and +16+hi*8.
static __device__ __forceinline__ v16bf load_frag(const __bf16* rowbase, int hi) {
  const v8bf a = *reinterpret_cast<const v8bf*>(rowbase + hi * 8);
  const v8bf b = *reinterpret_cast<const v8bf*>(rowbase + 16 + hi * 8);
  v16bf r;
#pragma unroll
  for (int t = 0; t < 8; ++t) { r[t] = a[t]; r[8 + t] = b[t]; }
  return r;
}

__global__ void ea_cast_bf16(const float* __restrict__ src, __bf16* __restrict__ dst, int n) {
  int i = blockIdx.x * blockDim.x + threadIdx.x;
  if (i < n) dst[i] = (__bf16)src[i];
}

// Wt[n*K + k] = W[k*N + n]  (store weights transposed so B-fragments are contiguous)
__global__ void ea_transpose_cast(const float* __restrict__ W, __bf16* __restrict__ Wt,
                                  int K, int N) {
  int idx = blockIdx.x * blockDim.x + threadIdx.x;
  if (idx >= K * N) return;
  int k = idx / N, n = idx % N;
  Wt[(size_t)n * K + k] = (__bf16)W[idx];
}

// Y = X(bf16, MxK) @ Wt^T(bf16, NoutxK rows) + bias.
// mode 0: bf16 out, (b, head, i, dd) layout   (Q, K)
// mode 1: bf16 out, (b, head, dd, i) layout   (V transposed)
// mode 2: f32 out, plain (M, Nout)            (final projection)
__global__ void __launch_bounds__(256)
ea_gemm_wmma(const __bf16* __restrict__ X, const __bf16* __restrict__ Wt,
             const float* __restrict__ bias, __bf16* __restrict__ Ybf,
             float* __restrict__ Yf, int Mrows, int Kd, int Nout, int mode) {
  const int lane   = threadIdx.x & 31;
  const int wave   = threadIdx.x >> 5;
  const int lanelo = lane & 15;
  const int hi     = lane >> 4;
  const int m0 = (blockIdx.y * 8 + wave) * 16;
  const int n0 = blockIdx.x * 16;
  if (m0 >= Mrows) return;  // wave-uniform

  v8f acc = {};
  const __bf16* arow = X  + (size_t)(m0 + lanelo) * Kd;
  const __bf16* brow = Wt + (size_t)(n0 + lanelo) * Kd;
  for (int k0 = 0; k0 < Kd; k0 += 32) {
    v16bf a = load_frag(arow + k0, hi);
    v16bf b = load_frag(brow + k0, hi);
    acc = __builtin_amdgcn_wmma_f32_16x16x32_bf16(false, a, false, b, (short)0, acc,
                                                  false, false);
  }
  const int n = n0 + lanelo;
  const float bn = bias[n];
#pragma unroll
  for (int r = 0; r < 8; ++r) {
    const int m = m0 + hi * 8 + r;            // C layout: row = hi*8+r, col = lanelo
    const float v = acc[r] + bn;
    if (mode == 2) {
      Yf[(size_t)m * Nout + n] = v;
    } else {
      const int bidx = m >> 10, i = m & 1023;
      const int head = n >> 6,  dd = n & 63;
      size_t o;
      if (mode == 0) o = (((size_t)(bidx * NHEADS + head)) * SEQ + i) * HD + dd;
      else           o = (((size_t)(bidx * NHEADS + head)) * HD + dd) * SEQ + i;
      Ybf[o] = (__bf16)v;
    }
  }
}

// Flash-attention with fused edge bias + dual mask.
// One wave = one 16-query tile; keys processed 32 at a time.
__global__ void __launch_bounds__(256)
ea_attn_wmma(const __bf16* __restrict__ Q, const __bf16* __restrict__ Kmat,
             const __bf16* __restrict__ Vt, const int* __restrict__ E,
             const int* __restrict__ Mask, const float* __restrict__ edge_emb,
             __bf16* __restrict__ Obf) {
  __shared__ float plds_all[8 * 16 * 32];  // per-wave 16x32 prob staging (16 KB)
  const int lane   = threadIdx.x & 31;
  const int wave   = threadIdx.x >> 5;
  const int lanelo = lane & 15;
  const int hi     = lane >> 4;
  const int bh   = blockIdx.y;
  const int b    = bh / NHEADS;
  const int head = bh % NHEADS;
  const int q0   = (blockIdx.x * 8 + wave) * 16;
  float* plds = plds_all + wave * 512;

  // Q fragments (16 queries x 64 dims = two K-chunks of 32)
  const __bf16* qrow = Q + ((size_t)bh * SEQ + q0 + lanelo) * HD;
  const v16bf qf0 = load_frag(qrow, hi);
  const v16bf qf1 = load_frag(qrow + 32, hi);

  int qm[8];
#pragma unroll
  for (int r = 0; r < 8; ++r) qm[r] = Mask[b * SEQ + q0 + hi * 8 + r];

  float mi[8], li[8];
#pragma unroll
  for (int r = 0; r < 8; ++r) { mi[r] = -INFINITY; li[r] = 0.0f; }
  v8f accO[4] = {};

  for (int j0 = 0; j0 < SEQ; j0 += 32) {
    float sv[2][8];
#pragma unroll
    for (int jh = 0; jh < 2; ++jh) {
      const int j = j0 + jh * 16 + lanelo;       // this lane's key column
      const __bf16* krow = Kmat + ((size_t)bh * SEQ + j) * HD;
      v8f s = {};
      v16bf kf0 = load_frag(krow, hi);
      s = __builtin_amdgcn_wmma_f32_16x16x32_bf16(false, qf0, false, kf0, (short)0, s,
                                                  false, false);
      v16bf kf1 = load_frag(krow + 32, hi);
      s = __builtin_amdgcn_wmma_f32_16x16x32_bf16(false, qf1, false, kf1, (short)0, s,
                                                  false, false);
      const int km = Mask[b * SEQ + j];
#pragma unroll
      for (int r = 0; r < 8; ++r) {
        const int i = q0 + hi * 8 + r;
        const int e = E[((size_t)b * SEQ + i) * SEQ + j];
        const float biasv = edge_emb[e * NHEADS + head];
        float val = s[r] * ATT_SCALE + biasv;
        if (!km || !qm[r]) val = -1e9f;
        sv[jh][r] = val;
      }
    }
    // online softmax: reduce across the 16 lanes holding each row
#pragma unroll
    for (int r = 0; r < 8; ++r) {
      float mx = fmaxf(sv[0][r], sv[1][r]);
#pragma unroll
      for (int d = 8; d >= 1; d >>= 1) mx = fmaxf(mx, __shfl_xor(mx, d, 32));
      const float mnew  = fmaxf(mi[r], mx);
      const float alpha = __expf(mi[r] - mnew);
      const float p0 = __expf(sv[0][r] - mnew);
      const float p1 = __expf(sv[1][r] - mnew);
      float ls = p0 + p1;
#pragma unroll
      for (int d = 8; d >= 1; d >>= 1) ls += __shfl_xor(ls, d, 32);
      li[r] = li[r] * alpha + ls;
      mi[r] = mnew;
#pragma unroll
      for (int t = 0; t < 4; ++t) accO[t][r] *= alpha;
      const int row = hi * 8 + r;                // C layout -> LDS (row, key)
      plds[row * 32 + lanelo]      = p0;
      plds[row * 32 + 16 + lanelo] = p1;
    }
    // our own LDS writes -> our own reads (same wave): explicit DS-counter wait
    asm volatile("s_wait_dscnt 0" ::: "memory");
    // rebuild P as an A-fragment (row within lane): row = lanelo, k per packing
    v16bf pa;
#pragma unroll
    for (int v = 0; v < 16; ++v) {
      const int kk = (v >> 3) * 16 + hi * 8 + (v & 7);
      pa[v] = (__bf16)plds[lanelo * 32 + kk];
    }
    // P(16x32) @ V(32keys x 64dims) via Vt rows (dim-major, keys contiguous)
#pragma unroll
    for (int t = 0; t < 4; ++t) {
      const __bf16* vrow = Vt + ((size_t)bh * HD + t * 16 + lanelo) * SEQ + j0;
      v16bf vf = load_frag(vrow, hi);
      accO[t] = __builtin_amdgcn_wmma_f32_16x16x32_bf16(false, pa, false, vf, (short)0,
                                                        accO[t], false, false);
    }
  }
  // normalize, store O in (b, i, h*64+dd) bf16 for the output projection
#pragma unroll
  for (int t = 0; t < 4; ++t) {
    const int n = t * 16 + lanelo;
#pragma unroll
    for (int r = 0; r < 8; ++r) {
      const int i = q0 + hi * 8 + r;
      const float o = accO[t][r] / li[r];
      Obf[((size_t)(b * SEQ + i)) * HDIM + head * HD + n] = (__bf16)o;
    }
  }
}

extern "C" void kernel_launch(void* const* d_in, const int* in_sizes, int n_in,
                              void* d_out, int out_size, void* d_ws, size_t ws_size,
                              hipStream_t stream) {
  const float* x   = (const float*)d_in[0];
  const int*   E   = (const int*)d_in[1];
  const int*   M   = (const int*)d_in[2];
  const float* Wq  = (const float*)d_in[3];
  const float* bq  = (const float*)d_in[4];
  const float* Wk  = (const float*)d_in[5];
  const float* bk  = (const float*)d_in[6];
  const float* Wv  = (const float*)d_in[7];
  const float* bv  = (const float*)d_in[8];
  const float* Wo  = (const float*)d_in[9];
  const float* bo  = (const float*)d_in[10];
  const float* eemb = (const float*)d_in[11];

  char* w = (char*)d_ws;
  auto carve = [&](size_t bytes) {
    char* p = w;
    w += (bytes + 255) & ~(size_t)255;
    return p;
  };
  const size_t xEl = (size_t)BATCH * SEQ * HDIM;          // 6291456
  const size_t wEl = (size_t)HDIM * HDIM;                 // 589824
  const size_t hEl = (size_t)BATCH * NHEADS * SEQ * HD;   // 6291456
  __bf16* xbf = (__bf16*)carve(xEl * 2);
  __bf16* wqt = (__bf16*)carve(wEl * 2);
  __bf16* wkt = (__bf16*)carve(wEl * 2);
  __bf16* wvt = (__bf16*)carve(wEl * 2);
  __bf16* wot = (__bf16*)carve(wEl * 2);
  __bf16* Qb  = (__bf16*)carve(hEl * 2);
  __bf16* Kb  = (__bf16*)carve(hEl * 2);
  __bf16* Vtb = (__bf16*)carve(hEl * 2);
  __bf16* Obf = (__bf16*)carve(xEl * 2);

  dim3 blk(256);
  ea_cast_bf16<<<dim3((unsigned)((xEl + 255) / 256)), blk, 0, stream>>>(x, xbf, (int)xEl);
  ea_transpose_cast<<<dim3((unsigned)((wEl + 255) / 256)), blk, 0, stream>>>(Wq, wqt, HDIM, HDIM);
  ea_transpose_cast<<<dim3((unsigned)((wEl + 255) / 256)), blk, 0, stream>>>(Wk, wkt, HDIM, HDIM);
  ea_transpose_cast<<<dim3((unsigned)((wEl + 255) / 256)), blk, 0, stream>>>(Wv, wvt, HDIM, HDIM);
  ea_transpose_cast<<<dim3((unsigned)((wEl + 255) / 256)), blk, 0, stream>>>(Wo, wot, HDIM, HDIM);

  dim3 gGemm(HDIM / 16, (BATCH * SEQ) / 128);  // 48 x 64 blocks, 8 waves each
  ea_gemm_wmma<<<gGemm, blk, 0, stream>>>(xbf, wqt, bq, Qb, nullptr,
                                          BATCH * SEQ, HDIM, HDIM, 0);
  ea_gemm_wmma<<<gGemm, blk, 0, stream>>>(xbf, wkt, bk, Kb, nullptr,
                                          BATCH * SEQ, HDIM, HDIM, 0);
  ea_gemm_wmma<<<gGemm, blk, 0, stream>>>(xbf, wvt, bv, Vtb, nullptr,
                                          BATCH * SEQ, HDIM, HDIM, 1);

  dim3 gAttn(SEQ / 128, BATCH * NHEADS);       // 8 x 96 blocks
  ea_attn_wmma<<<gAttn, blk, 0, stream>>>(Qb, Kb, Vtb, E, M, eemb, Obf);

  ea_gemm_wmma<<<gGemm, blk, 0, stream>>>(Obf, wot, bo, nullptr, (float*)d_out,
                                          BATCH * SEQ, HDIM, HDIM, 2);
}
